// AngleMessagePassing_34093450396330
// MI455X (gfx1250) — compile-verified, matrix operand order checked by reading, not run on hardware
//
#include <hip/hip_runtime.h>

#define N_NODES    100000
#define N_ANGLES   1000000
#define SCALAR_DIM 128
#define HIDDEN     64

typedef __attribute__((ext_vector_type(16))) _Float16 v16h;
typedef __attribute__((ext_vector_type(8)))  _Float16 v8h;
typedef __attribute__((ext_vector_type(8)))  float    v8f;
typedef __attribute__((ext_vector_type(4)))  float    v4f;

__device__ __forceinline__ v8f wmma16(v16h a, v16h b, v8f c) {
  // D = A(16x32 f16) x B(32x16 f16) + C(16x16 f32)
  return __builtin_amdgcn_wmma_f32_16x16x32_f16(
      /*neg_a=*/false, a, /*neg_b=*/false, b,
      /*c_mod=*/(short)0, c, /*reuse_a=*/false, /*reuse_b=*/false);
}

__device__ __forceinline__ float silu_f(float x) {
  return x * (1.0f / (1.0f + __expf(-x)));
}

// Build a 16-element f16 A-fragment from two 8-float runs (run1 = p0..p1,
// run2 = p2..p3), scaled by s. Matches the 16-bit A-matrix VGPR layout
// (lane half h: K = base..base+7 and base+16..base+23).
__device__ __forceinline__ v16h make_frag(v4f p0, v4f p1, v4f p2, v4f p3, float s) {
  v16h f;
  f[0]  = (_Float16)(p0[0]*s); f[1]  = (_Float16)(p0[1]*s);
  f[2]  = (_Float16)(p0[2]*s); f[3]  = (_Float16)(p0[3]*s);
  f[4]  = (_Float16)(p1[0]*s); f[5]  = (_Float16)(p1[1]*s);
  f[6]  = (_Float16)(p1[2]*s); f[7]  = (_Float16)(p1[3]*s);
  f[8]  = (_Float16)(p2[0]*s); f[9]  = (_Float16)(p2[1]*s);
  f[10] = (_Float16)(p2[2]*s); f[11] = (_Float16)(p2[3]*s);
  f[12] = (_Float16)(p3[0]*s); f[13] = (_Float16)(p3[1]*s);
  f[14] = (_Float16)(p3[2]*s); f[15] = (_Float16)(p3[3]*s);
  return f;
}

// ---------------------------------------------------------------------------
__global__ void __launch_bounds__(256) zero_kernel(float* __restrict__ p, int n) {
  int i = blockIdx.x * blockDim.x + threadIdx.x;
  if (i < n) p[i] = 0.0f;
}

// ---------------------------------------------------------------------------
// Pack an f32 row-major weight matrix W[Krows][Ncols] into f16 B-fragments:
// dst[((c*ntl + t)*32 + lane)*16 + e], lane = (n_in_tile | half<<4),
// e<8 -> k = c*32 + half*8 + e ; e>=8 -> k = c*32 + 16 + half*8 + (e-8).
// Out-of-range k (K padding) packs zeros.
// ---------------------------------------------------------------------------
__global__ void __launch_bounds__(256)
pack_weights(const float* __restrict__ W, _Float16* __restrict__ dst,
             int Krows, int Ncols, int nck, int ntl) {
  int gid = blockIdx.x * blockDim.x + threadIdx.x;
  int total = nck * ntl * 32;
  if (gid >= total) return;
  int lane = gid & 31;
  int t    = (gid >> 5) % ntl;
  int c    = (gid >> 5) / ntl;
  int nl   = lane & 15;
  int hh   = lane >> 4;
  int n    = t * 16 + nl;
  v16h f;
#pragma unroll
  for (int e = 0; e < 16; ++e) {
    int k = (e < 8) ? (c * 32 + hh * 8 + e)
                    : (c * 32 + 16 + hh * 8 + (e - 8));
    float v = (k < Krows && n < Ncols) ? W[(size_t)k * Ncols + n] : 0.0f;
    f[e] = (_Float16)v;
  }
  *(v16h*)(dst + (size_t)gid * 16) = f;
}

// ---------------------------------------------------------------------------
// Main per-angle MLP. One wave processes TWO 16-angle M-tiles (32 angles) so
// that each weight B-fragment load feeds two WMMAs:
//   in  = [x[i] | x[k] | attr]  (16 x 258, K padded to 288)
//   h1  = silu(in @ W1 + b1)    9 Kchunks x 4 Ntiles x 2 Mtiles WMMA
//   h2  = silu(h1 @ W2 + b2)    2 x 4 x 2 WMMA
//   msg = h2 @ W3 + b3          2 x 8 x 2 WMMA
//   atomicAdd into sums[j], cnts[j]  (L2-resident)
// ---------------------------------------------------------------------------
__global__ void __launch_bounds__(256)
mlp_kernel(const float* __restrict__ x,
           const int*   __restrict__ tri,     // (3, N_ANGLES): i, j, k
           const float* __restrict__ attr,    // (N_ANGLES, 2)
           const float* __restrict__ b1,
           const float* __restrict__ b2,
           const float* __restrict__ b3,
           const _Float16* __restrict__ W1pk, // 9 x 4 tiles
           const _Float16* __restrict__ W2pk, // 2 x 4 tiles
           const _Float16* __restrict__ W3pk, // 2 x 8 tiles
           float* __restrict__ sums,
           float* __restrict__ cnts) {
  __shared__ _Float16 hbuf[8][2][16][HIDDEN];   // per-wave 2x(16x64) slabs

  const int lane = threadIdx.x & 31;
  const int w    = threadIdx.x >> 5;
  const int wid  = blockIdx.x * 8 + w;
  const int a0   = wid * 32;                 // 2 M-tiles of 16 angles
  if (a0 >= N_ANGLES) return;                // uniform per wave

  const int m    = lane & 15;                // A-row owned by this lane
  const int h    = lane >> 4;                // lane half (K-run selector)
  const int off1 = h * 8;

  int am[2], iI[2], jI[2], kI[2];
  const float* rowI[2];
  const float* rowK[2];
  float at0[2], at1[2];
#pragma unroll
  for (int u = 0; u < 2; ++u) {
    am[u] = min(a0 + u * 16 + m, N_ANGLES - 1);
    iI[u] = tri[am[u]];
    jI[u] = tri[N_ANGLES + am[u]];
    kI[u] = tri[2 * N_ANGLES + am[u]];
    rowI[u] = x + (size_t)iI[u] * SCALAR_DIM;
    rowK[u] = x + (size_t)kI[u] * SCALAR_DIM;
    at0[u] = attr[(size_t)am[u] * 2 + 0];
    at1[u] = attr[(size_t)am[u] * 2 + 1];
  }

  // per-lane bias columns, hoisted
  float b1v[4], b2v[4], b3v[8];
#pragma unroll
  for (int t = 0; t < 4; ++t) { b1v[t] = b1[t * 16 + m]; b2v[t] = b2[t * 16 + m]; }
#pragma unroll
  for (int t = 0; t < 8; ++t) { b3v[t] = b3[t * 16 + m]; }

  // ----- layer 1: 9 K-chunks of 32, 4 N-tiles, 2 M-tiles -----
  v8f acc1[2][4] = {};
#pragma unroll
  for (int c = 0; c < 9; ++c) {
    v16h afrag[2];
#pragma unroll
    for (int u = 0; u < 2; ++u) {
      if (c < 8) {
        const float* src = (c < 4) ? rowI[u] : rowK[u];
        const int base = (c & 3) * 32 + off1;
        v4f p0 = *(const v4f*)(src + base);
        v4f p1 = *(const v4f*)(src + base + 4);
        v4f p2 = *(const v4f*)(src + base + 16);
        v4f p3 = *(const v4f*)(src + base + 20);
        afrag[u] = make_frag(p0, p1, p2, p3, 1.0f);
      } else {
        v16h zf = {};
        if (h == 0) { zf[0] = (_Float16)at0[u]; zf[1] = (_Float16)at1[u]; }
        afrag[u] = zf;                       // K=256..287: attr + zero pad
      }
    }
#pragma unroll
    for (int t = 0; t < 4; ++t) {
      v16h bfrag = *(const v16h*)(W1pk + ((size_t)(c * 4 + t) * 32 + lane) * 16);
      acc1[0][t] = wmma16(afrag[0], bfrag, acc1[0][t]);
      acc1[1][t] = wmma16(afrag[1], bfrag, acc1[1][t]);
    }
  }

  // silu + stage h1 into LDS (C/D layout -> row-major f16)
#pragma unroll
  for (int u = 0; u < 2; ++u)
#pragma unroll
    for (int t = 0; t < 4; ++t)
#pragma unroll
      for (int r = 0; r < 8; ++r)
        hbuf[w][u][h * 8 + r][t * 16 + m] =
            (_Float16)silu_f(acc1[u][t][r] + b1v[t]);

  // ----- layer 2: 2 K-chunks, 4 N-tiles, 2 M-tiles -----
  v8f acc2[2][4] = {};
#pragma unroll
  for (int c = 0; c < 2; ++c) {
    v16h afrag[2];
#pragma unroll
    for (int u = 0; u < 2; ++u) {
      v8h lo = *(const v8h*)&hbuf[w][u][m][c * 32 + off1];
      v8h hi = *(const v8h*)&hbuf[w][u][m][c * 32 + 16 + off1];
      afrag[u] = __builtin_shufflevector(lo, hi, 0, 1, 2, 3, 4, 5, 6, 7,
                                                 8, 9, 10, 11, 12, 13, 14, 15);
    }
#pragma unroll
    for (int t = 0; t < 4; ++t) {
      v16h bfrag = *(const v16h*)(W2pk + ((size_t)(c * 4 + t) * 32 + lane) * 16);
      acc2[0][t] = wmma16(afrag[0], bfrag, acc2[0][t]);
      acc2[1][t] = wmma16(afrag[1], bfrag, acc2[1][t]);
    }
  }

  // silu + stage h2 (reuse slabs; DS ops are in-order within a wave)
#pragma unroll
  for (int u = 0; u < 2; ++u)
#pragma unroll
    for (int t = 0; t < 4; ++t)
#pragma unroll
      for (int r = 0; r < 8; ++r)
        hbuf[w][u][h * 8 + r][t * 16 + m] =
            (_Float16)silu_f(acc2[u][t][r] + b2v[t]);

  // ----- layer 3: 2 K-chunks, 8 N-tiles, 2 M-tiles -----
  v8f acc3[2][8] = {};
#pragma unroll
  for (int c = 0; c < 2; ++c) {
    v16h afrag[2];
#pragma unroll
    for (int u = 0; u < 2; ++u) {
      v8h lo = *(const v8h*)&hbuf[w][u][m][c * 32 + off1];
      v8h hi = *(const v8h*)&hbuf[w][u][m][c * 32 + 16 + off1];
      afrag[u] = __builtin_shufflevector(lo, hi, 0, 1, 2, 3, 4, 5, 6, 7,
                                                 8, 9, 10, 11, 12, 13, 14, 15);
    }
#pragma unroll
    for (int t = 0; t < 8; ++t) {
      v16h bfrag = *(const v16h*)(W3pk + ((size_t)(c * 8 + t) * 32 + lane) * 16);
      acc3[0][t] = wmma16(afrag[0], bfrag, acc3[0][t]);
      acc3[1][t] = wmma16(afrag[1], bfrag, acc3[1][t]);
    }
  }

  // ----- scatter: msgs + b3 into sums[j], counts into cnts[j] -----
#pragma unroll
  for (int u = 0; u < 2; ++u) {
    int jrow[8];
#pragma unroll
    for (int r = 0; r < 8; ++r) jrow[r] = __shfl(jI[u], h * 8 + r, 32);
#pragma unroll
    for (int t = 0; t < 8; ++t)
#pragma unroll
      for (int r = 0; r < 8; ++r) {
        if (a0 + u * 16 + h * 8 + r < N_ANGLES) {
          atomicAdd(sums + (size_t)jrow[r] * SCALAR_DIM + t * 16 + m,
                    acc3[u][t][r] + b3v[t]);
        }
      }
    if (lane < 16 && (a0 + u * 16 + m) < N_ANGLES) atomicAdd(cnts + jI[u], 1.0f);
  }
}

// ---------------------------------------------------------------------------
// Finalize: aggr = sums / max(cnt,1);  out = aggr @ Wout.
// One wave per 16-node tile: 4 K-chunks x 8 N-tiles of WMMA.
// ---------------------------------------------------------------------------
__global__ void __launch_bounds__(256)
out_kernel(const float* __restrict__ sums,
           const float* __restrict__ cnts,
           const _Float16* __restrict__ Woutpk,  // 4 x 8 tiles
           float* __restrict__ out) {
  const int lane = threadIdx.x & 31;
  const int w    = threadIdx.x >> 5;
  const int n0   = (blockIdx.x * 8 + w) * 16;
  if (n0 >= N_NODES) return;                 // uniform per wave

  const int m = lane & 15;
  const int h = lane >> 4;
  const int node = min(n0 + m, N_NODES - 1);

  const float inv = 1.0f / fmaxf(cnts[node], 1.0f);
  const float* srow = sums + (size_t)node * SCALAR_DIM;

  v8f acc[8] = {};
#pragma unroll
  for (int c = 0; c < 4; ++c) {
    const int base = c * 32 + h * 8;
    v4f p0 = *(const v4f*)(srow + base);
    v4f p1 = *(const v4f*)(srow + base + 4);
    v4f p2 = *(const v4f*)(srow + base + 16);
    v4f p3 = *(const v4f*)(srow + base + 20);
    v16h afrag = make_frag(p0, p1, p2, p3, inv);
#pragma unroll
    for (int t = 0; t < 8; ++t) {
      v16h bfrag = *(const v16h*)(Woutpk + ((size_t)(c * 8 + t) * 32 + lane) * 16);
      acc[t] = wmma16(afrag, bfrag, acc[t]);
    }
  }

#pragma unroll
  for (int t = 0; t < 8; ++t)
#pragma unroll
    for (int r = 0; r < 8; ++r) {
      int row = n0 + h * 8 + r;
      if (row < N_NODES)
        out[(size_t)row * SCALAR_DIM + t * 16 + m] = acc[t][r];
    }
}

// ---------------------------------------------------------------------------
extern "C" void kernel_launch(void* const* d_in, const int* in_sizes, int n_in,
                              void* d_out, int out_size, void* d_ws, size_t ws_size,
                              hipStream_t stream) {
  const float* x    = (const float*)d_in[0];
  const int*   tri  = (const int*)d_in[1];
  const float* attr = (const float*)d_in[2];
  const float* W1   = (const float*)d_in[3];
  const float* b1   = (const float*)d_in[4];
  const float* W2   = (const float*)d_in[5];
  const float* b2   = (const float*)d_in[6];
  const float* W3   = (const float*)d_in[7];
  const float* b3   = (const float*)d_in[8];
  const float* Wout = (const float*)d_in[9];
  float* out = (float*)d_out;

  char* ws = (char*)d_ws;
  float* sums = (float*)ws;                                     // 51.2 MB
  float* cnts = (float*)(ws + (size_t)N_NODES * SCALAR_DIM * 4);// 0.4 MB
  size_t off = ((size_t)N_NODES * (SCALAR_DIM + 1) * 4 + 63) & ~(size_t)63;
  _Float16* W1pk   = (_Float16*)(ws + off); off += (size_t)9 * 4 * 32 * 16 * 2;
  _Float16* W2pk   = (_Float16*)(ws + off); off += (size_t)2 * 4 * 32 * 16 * 2;
  _Float16* W3pk   = (_Float16*)(ws + off); off += (size_t)2 * 8 * 32 * 16 * 2;
  _Float16* Woutpk = (_Float16*)(ws + off);

  const int nz = N_NODES * (SCALAR_DIM + 1);
  zero_kernel<<<(nz + 255) / 256, 256, 0, stream>>>(sums, nz);

  pack_weights<<<(9 * 4 * 32 + 255) / 256, 256, 0, stream>>>(W1, W1pk, 258, 64, 9, 4);
  pack_weights<<<(2 * 4 * 32 + 255) / 256, 256, 0, stream>>>(W2, W2pk, 64, 64, 2, 4);
  pack_weights<<<(2 * 8 * 32 + 255) / 256, 256, 0, stream>>>(W3, W3pk, 64, 128, 2, 8);
  pack_weights<<<(4 * 8 * 32 + 255) / 256, 256, 0, stream>>>(Wout, Woutpk, 128, 128, 4, 8);

  const int wtiles = (N_ANGLES + 31) / 32;          // 31,250 wave-tiles
  mlp_kernel<<<(wtiles + 7) / 8, 256, 0, stream>>>(x, tri, attr, b1, b2, b3,
                                                   W1pk, W2pk, W3pk, sums, cnts);

  const int otiles = (N_NODES + 15) / 16;           // 6,250
  out_kernel<<<(otiles + 7) / 8, 256, 0, stream>>>(sums, cnts, Woutpk, out);
}
